// GCNLayer_9423158248174
// MI455X (gfx1250) — matrix operand order repaired and checked by reading, not compiled, not run on hardware
//
#include <hip/hip_runtime.h>
#include <hip/hip_bf16.h>

// ---------------------------------------------------------------------------
// GCN layer on MI455X (gfx1250, wave32):
//   out = diag(dn) * ( A @ ( diag(dn) * (x @ W) ) ) + b,  dn = rsqrt(rowsum(A)+1e-8)
// Big GEMM (A:2048x2048 fp32 streamed from HBM) is memory-bound at 23.3TB/s;
// fp32 -> bf16 hi/lo split + 3x v_wmma_f32_16x16x32_bf16 (precision-compensated)
// keeps matrix compute hidden under HBM traffic at ~fp32 accuracy.
// Kernel 2 pre-splits and pre-transposes xw into bf16 hi/lo [b][col][m] arrays
// (one b128 store per lane, straight from the WMMA D-fragment layout), so the
// hot aggregation loop stages B with pure aligned b128 copies: no conversion,
// no scalar transpose stores inside the 64x-replayed k-loop.
// LDS row pitch = 40 bf16 (80B): 16B-aligned rows -> all fragment reads are
// ds_load_b128, and the 20-bank row stride is conflict-free across 16 lanes.
// ---------------------------------------------------------------------------

typedef __attribute__((ext_vector_type(2)))  float  f32x2;
typedef __attribute__((ext_vector_type(8)))  float  f32x8;
typedef __attribute__((ext_vector_type(16))) __bf16 bf16x16;

#define NB   8
#define NN   2048
#define DF   128
#define TKC  32            // k-chunk for aggregation GEMM
#define LDA  40            // padded LDS row stride (bf16): 80B rows, 16B aligned

union FragCast { uint4 q[2]; bf16x16 v; };     // two ds_load_b128 -> one fragment
union Bf2Pack  { __bf16 h[2]; unsigned int u; };
union Bf8Pack  { __bf16 h[8]; uint4 q; };

// ---------------------------------------------------------------------------
// Kernel 1: dn[b,n] = rsqrt(sum_m adj[b,n,m] + 1e-8). One wave per row.
// ---------------------------------------------------------------------------
__global__ __launch_bounds__(256) void gcn_degree(const float* __restrict__ adj,
                                                  float* __restrict__ dn) {
  const int row  = blockIdx.x * 8 + (threadIdx.x >> 5);   // flat over B*N
  const int lane = threadIdx.x & 31;
  const float4* p = reinterpret_cast<const float4*>(adj + (size_t)row * NN);
  float s = 0.f;
#pragma unroll
  for (int i = 0; i < NN / 4 / 32; ++i) {                 // 16 float4 per lane
    float4 v = p[lane + 32 * i];
    s += v.x + v.y + v.z + v.w;
  }
#pragma unroll
  for (int off = 16; off > 0; off >>= 1) s += __shfl_xor(s, off, 32);
  if (lane == 0) dn[row] = rsqrtf(s + 1e-8f);
}

// ---------------------------------------------------------------------------
// Kernel 2: t = dn[r] * (x[r,:] @ W) via exact fp32 v_wmma_f32_16x16x4_f32,
// then split t into bf16 hi/lo and store TRANSPOSED: xwT[b][col][m].
// D-fragment: lane holds col = col0+lane%16, rows m0..m0+7 -> one b128 store.
// ---------------------------------------------------------------------------
__global__ __launch_bounds__(256) void gcn_xw(const float* __restrict__ x,
                                              const float* __restrict__ W,
                                              const float* __restrict__ dn,
                                              __bf16* __restrict__ xwT_hi,
                                              __bf16* __restrict__ xwT_lo) {
  const int row0 = blockIdx.x * 16;          // flat row in [0, B*N)
  const int wave = threadIdx.x >> 5;
  const int lane = threadIdx.x & 31;
  const int col0 = wave * 16;
  const int l15  = lane & 15;
  const int hsel = lane >> 4;                // 0: lanes 0-15, 1: lanes 16-31

  const float* xrow = x + (size_t)(row0 + l15) * DF;   // A row for this lane
  f32x8 acc = {};
#pragma unroll 4
  for (int k = 0; k < DF; k += 4) {
    const int ka = k + hsel * 2;             // A: VGPR0=K0,1 (lo lanes) / K2,3 (hi lanes)
    f32x2 a, b;
    a[0] = xrow[ka];
    a[1] = xrow[ka + 1];
    b[0] = W[(size_t)ka * DF + col0 + l15];  // B: lanes 0-15 K0,1; lanes 16-31 K2,3
    b[1] = W[(size_t)(ka + 1) * DF + col0 + l15];
    acc = __builtin_amdgcn_wmma_f32_16x16x4_f32(false, a, false, b,
                                                (short)0, acc, false, false);
  }
  // Epilogue: scale by dn, split to bf16 hi/lo, store transposed (8 consec. m)
  const int bq = row0 >> 11;                 // batch index (NN = 2048 rows/batch)
  const int m0 = (row0 & (NN - 1)) + 8 * hsel;
  Bf8Pack ph, pl;
#pragma unroll
  for (int v = 0; v < 8; ++v) {
    const float s = dn[row0 + v + 8 * hsel] * acc[v];
    const __bf16 h = (__bf16)s;
    ph.h[v] = h;
    pl.h[v] = (__bf16)(s - (float)h);
  }
  const size_t base = ((size_t)bq * DF + col0 + l15) * NN + m0;   // 16B aligned
  *reinterpret_cast<uint4*>(xwT_hi + base) = ph.q;
  *reinterpret_cast<uint4*>(xwT_lo + base) = pl.q;
}

// ---------------------------------------------------------------------------
// Kernel 3: out[b,n,:] = dn[b,n] * (A[b] @ xw[b])[n,:] + bias
// Workgroup: 64 rows x 128 cols. 8 waves, wave (r=w>>1, c=w&1) owns 16x64.
// Double-buffered LDS. A: fp32 -> bf16 hi/lo split on the fly.
// B: pure aligned b128 copies from pre-split, pre-transposed xwT arrays.
// ---------------------------------------------------------------------------
__global__ __launch_bounds__(256) void gcn_aggregate(const float* __restrict__ adj,
                                                     const __bf16* __restrict__ xwT_hi,
                                                     const __bf16* __restrict__ xwT_lo,
                                                     const float* __restrict__ dnorm,
                                                     const float* __restrict__ bias,
                                                     float* __restrict__ out) {
  const int b  = blockIdx.x >> 5;            // 32 row-blocks of 64 per batch
  const int n0 = (blockIdx.x & 31) * 64;

  __shared__ __align__(16) __bf16 Ahi[2][64 * LDA];
  __shared__ __align__(16) __bf16 Alo[2][64 * LDA];
  __shared__ __align__(16) __bf16 Bhi[2][DF * LDA];   // [col][k]
  __shared__ __align__(16) __bf16 Blo[2][DF * LDA];

  const float*  adjB   = adj    + (size_t)b * NN * NN;
  const __bf16* xwThiB = xwT_hi + (size_t)b * DF * NN;
  const __bf16* xwTloB = xwT_lo + (size_t)b * DF * NN;

  const int tid  = threadIdx.x;
  const int wave = tid >> 5;
  const int lane = tid & 31;
  const int wr   = (wave >> 1) * 16;         // row strip within 64-block
  const int wc   = (wave & 1) * 64;          // col half
  const int l15  = lane & 15;
  const int hsel = lane >> 4;

  auto loadTile = [&](int kk, int bufw) {
    // adj tile: 64 rows x 32 cols, fp32 -> bf16 hi/lo; packed 8B LDS stores
#pragma unroll
    for (int i = 0; i < 2; ++i) {
      const int idx = tid + i * 256;         // 512 float4 total
      const int r   = idx >> 3;
      const int c   = (idx & 7) << 2;
      float4 v = *reinterpret_cast<const float4*>(adjB + (size_t)(n0 + r) * NN + kk + c);
      const float vv[4] = {v.x, v.y, v.z, v.w};
      Bf2Pack ph[2], pl[2];
#pragma unroll
      for (int j = 0; j < 4; ++j) {
        const __bf16 h = (__bf16)vv[j];
        ph[j >> 1].h[j & 1] = h;
        pl[j >> 1].h[j & 1] = (__bf16)(vv[j] - (float)h);
      }
      *reinterpret_cast<uint2*>(&Ahi[bufw][r * LDA + c]) = make_uint2(ph[0].u, ph[1].u);
      *reinterpret_cast<uint2*>(&Alo[bufw][r * LDA + c]) = make_uint2(pl[0].u, pl[1].u);
    }
    // xwT tiles: [128 cols][32 k] bf16, contiguous in k -> straight b128 copies
    {
      const int colc = tid >> 1;             // 0..127
      const int kh   = (tid & 1) * 16;       // k-offset 0 or 16
      const size_t g = (size_t)colc * NN + kk + kh;      // 16B aligned
      const uint4* gh = reinterpret_cast<const uint4*>(xwThiB + g);
      const uint4* gl = reinterpret_cast<const uint4*>(xwTloB + g);
      const uint4 h0 = gh[0], h1 = gh[1], l0 = gl[0], l1 = gl[1];
      *reinterpret_cast<uint4*>(&Bhi[bufw][colc * LDA + kh])     = h0;
      *reinterpret_cast<uint4*>(&Bhi[bufw][colc * LDA + kh + 8]) = h1;
      *reinterpret_cast<uint4*>(&Blo[bufw][colc * LDA + kh])     = l0;
      *reinterpret_cast<uint4*>(&Blo[bufw][colc * LDA + kh + 8]) = l1;
    }
  };

  f32x8 acc[4] = {};

  loadTile(0, 0);
  __syncthreads();

#pragma unroll 2
  for (int it = 0; it < NN / TKC; ++it) {
    const int buf = it & 1;
    if (it + 1 < NN / TKC) loadTile((it + 1) * TKC, buf ^ 1);   // prefetch next chunk

    // A fragment (16x32 bf16): lane<16: K0-7 & K16-23; lane>=16: K8-15 & K24-31
    const int arow = wr + l15;
    const int kbA  = hsel * 8;
    FragCast fah, fal;
    fah.q[0] = *reinterpret_cast<const uint4*>(&Ahi[buf][arow * LDA + kbA]);
    fah.q[1] = *reinterpret_cast<const uint4*>(&Ahi[buf][arow * LDA + 16 + kbA]);
    fal.q[0] = *reinterpret_cast<const uint4*>(&Alo[buf][arow * LDA + kbA]);
    fal.q[1] = *reinterpret_cast<const uint4*>(&Alo[buf][arow * LDA + 16 + kbA]);
    const bf16x16 ah = fah.v, al = fal.v;

    const int kbB = hsel * 8;                // B: lanes 0-15 K0-7 & K16-23; lanes 16-31 K8-15 & K24-31
#pragma unroll
    for (int t = 0; t < 4; ++t) {
      const int bcol = wc + t * 16 + l15;
      FragCast fbh, fbl;
      fbh.q[0] = *reinterpret_cast<const uint4*>(&Bhi[buf][bcol * LDA + kbB]);
      fbh.q[1] = *reinterpret_cast<const uint4*>(&Bhi[buf][bcol * LDA + kbB + 16]);
      fbl.q[0] = *reinterpret_cast<const uint4*>(&Blo[buf][bcol * LDA + kbB]);
      fbl.q[1] = *reinterpret_cast<const uint4*>(&Blo[buf][bcol * LDA + kbB + 16]);
      // precision-compensated bf16x3: a*b ~= ah*bh + al*bh + ah*bl
      acc[t] = __builtin_amdgcn_wmma_f32_16x16x32_bf16(false, ah, false, fbh.v,
                                                       (short)0, acc[t], false, false);
      acc[t] = __builtin_amdgcn_wmma_f32_16x16x32_bf16(false, al, false, fbh.v,
                                                       (short)0, acc[t], false, false);
      acc[t] = __builtin_amdgcn_wmma_f32_16x16x32_bf16(false, ah, false, fbl.v,
                                                       (short)0, acc[t], false, false);
    }
    __syncthreads();
  }

  // Epilogue: out = dn[row] * acc + bias[col]
  float dnv[8];
#pragma unroll
  for (int v = 0; v < 8; ++v)
    dnv[v] = dnorm[(size_t)b * NN + n0 + wr + v + 8 * hsel];
#pragma unroll
  for (int t = 0; t < 4; ++t) {
    const int col = wc + t * 16 + l15;
    const float bv = bias[col];
#pragma unroll
    for (int v = 0; v < 8; ++v) {
      const int row = n0 + wr + v + 8 * hsel;
      out[((size_t)b * NN + row) * DF + col] = dnv[v] * acc[t][v] + bv;
    }
  }
}

// ---------------------------------------------------------------------------
extern "C" void kernel_launch(void* const* d_in, const int* in_sizes, int n_in,
                              void* d_out, int out_size, void* d_ws, size_t ws_size,
                              hipStream_t stream) {
  (void)in_sizes; (void)n_in; (void)out_size; (void)ws_size;
  const float* x    = (const float*)d_in[0];   // [B,N,DIN]
  const float* adj  = (const float*)d_in[1];   // [B,N,N]
  const float* W    = (const float*)d_in[2];   // [DIN,DOUT]
  const float* bias = (const float*)d_in[3];   // [DOUT]
  float* out = (float*)d_out;                  // [B,N,DOUT]

  float*  dn     = (float*)d_ws;                       // B*N floats   (64 KB)
  __bf16* xwT_hi = (__bf16*)(dn + (size_t)NB * NN);    // B*DF*NN bf16 (4 MB)
  __bf16* xwT_lo = xwT_hi + (size_t)NB * DF * NN;      // B*DF*NN bf16 (4 MB)

  gcn_degree   <<<NB * NN / 8, 256, 0, stream>>>(adj, dn);
  gcn_xw       <<<NB * NN / 16, 256, 0, stream>>>(x, W, dn, xwT_hi, xwT_lo);
  gcn_aggregate<<<NB * (NN / 64), 256, 0, stream>>>(adj, xwT_hi, xwT_lo, dn, bias, out);
}
